// EfficientNetB0_BiGRU_MambaSSM_83580063580382
// MI455X (gfx1250) — matrix operand order repaired and checked by reading, not compiled, not run on hardware
//
#include <hip/hip_runtime.h>
#include <math.h>

typedef __attribute__((ext_vector_type(16))) _Float16 v16h;
typedef __attribute__((ext_vector_type(8)))  float    v8f;

#define BSZ  32
#define TLEN 10
#define FIN  1280
#define FDIM 256
#define GH   128
#define MH   256

// ---------------- WMMA fragment helpers (wave32, 16x16x32 f16 -> f32) ----------------

// A fragment: 16x32 tile of row-major X (ld = row stride). f32 -> f16 convert on load.
// lane<16: row=lane, K = k0 + {0..7, 16..23}; lane>=16: row=lane-16, K = k0 + {8..15, 24..31}
__device__ __forceinline__ v16h frag_a_f32(const float* X, int ld, int row0, int k0) {
  int lane = threadIdx.x & 31;
  const float* p = X + (size_t)(row0 + (lane & 15)) * ld + (k0 + ((lane >> 4) << 3));
  v16h a;
#pragma unroll
  for (int e = 0; e < 8; ++e) {
    a[e]     = (_Float16)p[e];
    a[e + 8] = (_Float16)p[e + 16];
  }
  return a;
}

// B fragment for C = X * W^T with W row-major (N x K): B[k][n] = W[n][k].
// lane<16: col n = lane, K = k0 + {0..7, 16..23} ; lane>=16: col n = lane-16, K = k0 + {8..15, 24..31}
__device__ __forceinline__ v16h frag_b_f32(const float* W, int ld, int n0, int k0) {
  int lane = threadIdx.x & 31;
  const float* p = W + (size_t)(n0 + (lane & 15)) * ld + (k0 + ((lane >> 4) << 3));
  v16h b;
#pragma unroll
  for (int e = 0; e < 8; ++e) {
    b[e]     = (_Float16)p[e];
    b[e + 8] = (_Float16)p[e + 16];
  }
  return b;
}

__device__ __forceinline__ v8f wmma_acc(v16h a, v16h b, v8f c) {
  return __builtin_amdgcn_wmma_f32_16x16x32_f16(false, a, false, b, (short)0, c, false, false);
}

__device__ __forceinline__ float wave_sum(float s) {
#pragma unroll
  for (int m = 16; m >= 1; m >>= 1) s += __shfl_xor(s, m, 32);
  return s;
}

__device__ __forceinline__ float sigmoidf_(float x) { return 1.0f / (1.0f + __expf(-x)); }

// ---------------- 1) adaptive avg pool: (320,1280,7,7) -> (320,1280) ----------------
// one wave per (n,c): 49 contiguous floats, wave-reduce. HBM-bound stage (80 MB read).
__global__ void pool49_kernel(const float* __restrict__ feats, float* __restrict__ pooled) {
  int wid  = (blockIdx.x * blockDim.x + threadIdx.x) >> 5;
  int lane = threadIdx.x & 31;
  const float* p = feats + (size_t)wid * 49;
  float s = p[lane];
  if (lane < 17) s += p[lane + 32];
  s = wave_sum(s);
  if (lane == 0) pooled[wid] = s * (1.0f / 49.0f);
}

// ---------------- 2) proj + BN(eval) + ReLU: x = relu(BN(pooled @ proj_w^T + b)) ----
// one wave per 16x16 output tile; M=320, N=256, K=1280
__global__ void proj_bn_relu_kernel(const float* __restrict__ X, const float* __restrict__ W,
                                    const float* __restrict__ bias,
                                    const float* __restrict__ bn_g, const float* __restrict__ bn_b,
                                    const float* __restrict__ bn_m, const float* __restrict__ bn_v,
                                    float* __restrict__ C) {
  const int N = FDIM, K = FIN;
  int wid = (blockIdx.x * blockDim.x + threadIdx.x) >> 5;
  int tn = wid % (N >> 4), tm = wid / (N >> 4);
  v8f acc = {};
  for (int k0 = 0; k0 < K; k0 += 32) {
    v16h a = frag_a_f32(X, K, tm * 16, k0);
    v16h b = frag_b_f32(W, K, tn * 16, k0);
    acc = wmma_acc(a, b, acc);
  }
  int lane = threadIdx.x & 31;
  int col  = tn * 16 + (lane & 15);
  int row0 = tm * 16 + ((lane >> 4) << 3);
  float bv = bias[col];
  float sc = __frsqrt_rn(bn_v[col] + 1e-5f) * bn_g[col];
  float sh = bn_b[col] - bn_m[col] * sc;
#pragma unroll
  for (int i = 0; i < 8; ++i) {
    float v = (acc[i] + bv) * sc + sh;
    C[(size_t)(row0 + i) * N + col] = v > 0.0f ? v : 0.0f;
  }
}

// ---------------- generic C = X @ W^T (+bias), all dims multiple of 16/32 ----------
__global__ void gemm_bias_kernel(const float* __restrict__ X, const float* __restrict__ W,
                                 const float* __restrict__ bias, float* __restrict__ C,
                                 int N, int K) {
  int wid = (blockIdx.x * blockDim.x + threadIdx.x) >> 5;
  int tn = wid % (N >> 4), tm = wid / (N >> 4);
  v8f acc = {};
  for (int k0 = 0; k0 < K; k0 += 32) {
    v16h a = frag_a_f32(X, K, tm * 16, k0);
    v16h b = frag_b_f32(W, K, tn * 16, k0);
    acc = wmma_acc(a, b, acc);
  }
  int lane = threadIdx.x & 31;
  int col  = tn * 16 + (lane & 15);
  int row0 = tm * 16 + ((lane >> 4) << 3);
  float bv = bias ? bias[col] : 0.0f;
#pragma unroll
  for (int i = 0; i < 8; ++i) C[(size_t)(row0 + i) * N + col] = acc[i] + bv;
}

// ---------------- 3) persistent BiGRU: grid.x = dir (0 fwd, 1 rev), 512 threads ----
// LDS-resident h (32x128). Weight B-fragments of whh are time-invariant: cached in
// VGPRs across all 10 steps (12 x v16h per wave). Per step: 4 LDS A-frags (shared by
// the wave's 3 column tiles) + 12 WMMAs, then fused gates. Next step's gi prefetched.
__global__ void __launch_bounds__(512) gru_kernel(const float* __restrict__ gi_f,
                                                  const float* __restrict__ gi_r,
                                                  const float* __restrict__ whh_f,
                                                  const float* __restrict__ whh_r,
                                                  const float* __restrict__ bhh_f,
                                                  const float* __restrict__ bhh_r,
                                                  float* __restrict__ x2) {
  __shared__ float sh_h[BSZ * GH];        // 16 KB
  __shared__ float sh_g[BSZ * 3 * GH];    // 48 KB
  const int dir = blockIdx.x;
  const float* gi  = dir ? gi_r  : gi_f;
  const float* whh = dir ? whh_r : whh_f;
  const float* bhh = dir ? bhh_r : bhh_f;
  const int tid = threadIdx.x, wave = tid >> 5, lane = tid & 31;
  const int tm = wave & 1;                // same row-tile for all 3 reps of this wave

  // time-invariant B fragments, resident in VGPRs for the whole recurrence
  v16h bf[3][4];
#pragma unroll
  for (int rep = 0; rep < 3; ++rep) {
    int tn = (wave >> 1) + 8 * rep;
#pragma unroll
    for (int kc = 0; kc < 4; ++kc) bf[rep][kc] = frag_b_f32(whh, GH, tn * 16, kc * 32);
  }

  for (int i = tid; i < BSZ * GH; i += 512) sh_h[i] = 0.0f;
  __syncthreads();

#pragma unroll 1
  for (int s = 0; s < TLEN; ++s) {
    const int tq = dir ? (TLEN - 1 - s) : s;
    // prefetch next step's gi rows (32 rows x 1536 B) while this step computes
    if (s + 1 < TLEN) {
      const int tq2 = dir ? (TLEN - 2 - s) : (s + 1);
      for (int i = tid; i < BSZ * 24; i += 512) {
        int b = i / 24, l = i % 24;
        __builtin_prefetch(gi + (size_t)(b * TLEN + tq2) * (3 * GH) + l * 16, 0, 0);
      }
    }
    // gh(32x384) = h(32x128) @ whh^T : A-frags loaded once, reused by 3 col tiles
    v16h af[4];
#pragma unroll
    for (int kc = 0; kc < 4; ++kc) af[kc] = frag_a_f32(sh_h, GH, tm * 16, kc * 32);
    v8f acc[3] = {{}, {}, {}};
#pragma unroll
    for (int rep = 0; rep < 3; ++rep)
#pragma unroll
      for (int kc = 0; kc < 4; ++kc) acc[rep] = wmma_acc(af[kc], bf[rep][kc], acc[rep]);
#pragma unroll
    for (int rep = 0; rep < 3; ++rep) {
      int tn = (wave >> 1) + 8 * rep;
      int col  = tn * 16 + (lane & 15);
      int row0 = tm * 16 + ((lane >> 4) << 3);
#pragma unroll
      for (int i = 0; i < 8; ++i) sh_g[(row0 + i) * (3 * GH) + col] = acc[rep][i];
    }
    __syncthreads();
    // gates
    for (int idx = tid; idx < BSZ * GH; idx += 512) {
      int b = idx >> 7, j = idx & (GH - 1);
      const float* gt = gi + (size_t)(b * TLEN + tq) * (3 * GH);
      float i_r = gt[j], i_z = gt[GH + j], i_n = gt[2 * GH + j];
      const float* gh = sh_g + b * (3 * GH);
      float h_r = gh[j] + bhh[j];
      float h_z = gh[GH + j] + bhh[GH + j];
      float h_n = gh[2 * GH + j] + bhh[2 * GH + j];
      float r = sigmoidf_(i_r + h_r);
      float z = sigmoidf_(i_z + h_z);
      float n = tanhf(i_n + r * h_n);
      float h = sh_h[idx];
      float hn = (1.0f - z) * n + z * h;
      sh_h[idx] = hn;
      x2[(size_t)(b * TLEN + tq) * (2 * GH) + dir * GH + j] = hn;
    }
    __syncthreads();
  }
}

// ---------------- 4) persistent SSM + LN + fc: single block, 512 threads ----------
// per step: hp = h@A^T + u_t ; h = gelu(hp) ; y = h@Cm^T ; out = LN(y)@fc_w^T + fc_b
// A-frags loaded once per GEMM and reused by the wave's 2 column tiles (same tm).
__global__ void __launch_bounds__(512) mamba_kernel(const float* __restrict__ Amat,
                                                    const float* __restrict__ Cm,
                                                    const float* __restrict__ u,
                                                    const float* __restrict__ ln_g,
                                                    const float* __restrict__ ln_b,
                                                    const float* __restrict__ fc_w,
                                                    const float* __restrict__ fc_b,
                                                    float* __restrict__ out) {
  __shared__ float sh_h[BSZ * MH];  // 32 KB
  __shared__ float sh_y[BSZ * MH];  // 32 KB
  const int tid = threadIdx.x, wave = tid >> 5, lane = tid & 31;
  const int tm = wave & 1;          // same row-tile for both reps of this wave

  for (int i = tid; i < BSZ * MH; i += 512) sh_h[i] = 0.0f;
  __syncthreads();

#pragma unroll 1
  for (int t = 0; t < TLEN; ++t) {
    // prefetch next step's u rows (32 x 1024 B)
    if (t + 1 < TLEN) {
      for (int i = tid; i < BSZ * 16; i += 512) {
        int b = i >> 4, l = i & 15;
        __builtin_prefetch(u + (size_t)(b * TLEN + t + 1) * MH + l * 16, 0, 0);
      }
    }
    // GEMM1: sh_y = h @ A^T  (32x256, K=256) : 32 tiles, 16 waves x 2 col tiles
    {
      v16h af[8];
#pragma unroll
      for (int kc = 0; kc < 8; ++kc) af[kc] = frag_a_f32(sh_h, MH, tm * 16, kc * 32);
      v8f acc[2] = {{}, {}};
#pragma unroll
      for (int rep = 0; rep < 2; ++rep) {
        int tn = (wave >> 1) + 8 * rep;
#pragma unroll
        for (int kc = 0; kc < 8; ++kc)
          acc[rep] = wmma_acc(af[kc], frag_b_f32(Amat, MH, tn * 16, kc * 32), acc[rep]);
      }
#pragma unroll
      for (int rep = 0; rep < 2; ++rep) {
        int tn = (wave >> 1) + 8 * rep;
        int col  = tn * 16 + (lane & 15);
        int row0 = tm * 16 + ((lane >> 4) << 3);
#pragma unroll
        for (int i = 0; i < 8; ++i) sh_y[(row0 + i) * MH + col] = acc[rep][i];
      }
    }
    __syncthreads();
    // exact GELU(h_pre + u_t) -> sh_h
    for (int idx = tid; idx < BSZ * MH; idx += 512) {
      int b = idx >> 8, j = idx & (MH - 1);
      float v = sh_y[idx] + u[(size_t)(b * TLEN + t) * MH + j];
      sh_h[idx] = 0.5f * v * (1.0f + erff(v * 0.70710678118654752f));
    }
    __syncthreads();
    // GEMM2: sh_y = h @ Cm^T
    {
      v16h af[8];
#pragma unroll
      for (int kc = 0; kc < 8; ++kc) af[kc] = frag_a_f32(sh_h, MH, tm * 16, kc * 32);
      v8f acc[2] = {{}, {}};
#pragma unroll
      for (int rep = 0; rep < 2; ++rep) {
        int tn = (wave >> 1) + 8 * rep;
#pragma unroll
        for (int kc = 0; kc < 8; ++kc)
          acc[rep] = wmma_acc(af[kc], frag_b_f32(Cm, MH, tn * 16, kc * 32), acc[rep]);
      }
#pragma unroll
      for (int rep = 0; rep < 2; ++rep) {
        int tn = (wave >> 1) + 8 * rep;
        int col  = tn * 16 + (lane & 15);
        int row0 = tm * 16 + ((lane >> 4) << 3);
#pragma unroll
        for (int i = 0; i < 8; ++i) sh_y[(row0 + i) * MH + col] = acc[rep][i];
      }
    }
    __syncthreads();
    // LayerNorm + fc: 32 rows, one wave handles 2 rows
#pragma unroll
    for (int rr = 0; rr < 2; ++rr) {
      int b = wave * 2 + rr;
      const float* yr = sh_y + b * MH;
      float v[8];
      float s = 0.0f;
#pragma unroll
      for (int e = 0; e < 8; ++e) { v[e] = yr[lane + 32 * e]; s += v[e]; }
      s = wave_sum(s);
      float mu = s * (1.0f / MH);
      float q = 0.0f;
#pragma unroll
      for (int e = 0; e < 8; ++e) { float d = v[e] - mu; q += d * d; }
      q = wave_sum(q);
      float inv = __frsqrt_rn(q * (1.0f / MH) + 1e-5f);
      float a0 = 0.0f, a1 = 0.0f;
#pragma unroll
      for (int e = 0; e < 8; ++e) {
        int j = lane + 32 * e;
        float nv = (v[e] - mu) * inv * ln_g[j] + ln_b[j];
        a0 += nv * fc_w[j];
        a1 += nv * fc_w[MH + j];
      }
      a0 = wave_sum(a0);
      a1 = wave_sum(a1);
      if (lane == 0) {
        out[(size_t)(b * TLEN + t) * 2 + 0] = a0 + fc_b[0];
        out[(size_t)(b * TLEN + t) * 2 + 1] = a1 + fc_b[1];
      }
    }
    __syncthreads();
  }
}

// ---------------- host launch ----------------
extern "C" void kernel_launch(void* const* d_in, const int* in_sizes, int n_in,
                              void* d_out, int out_size, void* d_ws, size_t ws_size,
                              hipStream_t stream) {
  const float* feats  = (const float*)d_in[0];
  // d_in[1] = seq_len (int, == 10, baked into constants)
  const float* proj_w = (const float*)d_in[2];
  const float* proj_b = (const float*)d_in[3];
  const float* bn_g   = (const float*)d_in[4];
  const float* bn_b   = (const float*)d_in[5];
  const float* bn_m   = (const float*)d_in[6];
  const float* bn_v   = (const float*)d_in[7];
  const float* wih_f  = (const float*)d_in[8];
  const float* whh_f  = (const float*)d_in[9];
  const float* bih_f  = (const float*)d_in[10];
  const float* bhh_f  = (const float*)d_in[11];
  const float* wih_r  = (const float*)d_in[12];
  const float* whh_r  = (const float*)d_in[13];
  const float* bih_r  = (const float*)d_in[14];
  const float* bhh_r  = (const float*)d_in[15];
  const float* Amat   = (const float*)d_in[16];
  const float* Bm     = (const float*)d_in[17];
  const float* Cm     = (const float*)d_in[18];
  const float* ln_g   = (const float*)d_in[19];
  const float* ln_b   = (const float*)d_in[20];
  const float* fc_w   = (const float*)d_in[21];
  const float* fc_b   = (const float*)d_in[22];
  float* out = (float*)d_out;

  const int M = BSZ * TLEN;  // 320
  float* ws     = (float*)d_ws;
  float* pooled = ws;                        // 320*1280
  float* x      = pooled + (size_t)M * FIN;  // 320*256
  float* gi_f   = x + (size_t)M * FDIM;      // 320*384
  float* gi_r   = gi_f + (size_t)M * 3 * GH; // 320*384
  float* x2     = gi_r + (size_t)M * 3 * GH; // 320*256
  float* u      = x2 + (size_t)M * (2 * GH); // 320*256

  // 1) pool: 320*1280 waves, 8 waves/block
  pool49_kernel<<<(M * FIN) / 8, 256, 0, stream>>>(feats, pooled);

  // 2) proj + BN + ReLU: (320x256, K=1280) -> 320 tiles -> 40 blocks
  proj_bn_relu_kernel<<<(M / 16) * (FDIM / 16) / 8, 256, 0, stream>>>(
      pooled, proj_w, proj_b, bn_g, bn_b, bn_m, bn_v, x);

  // 3) gi = x @ wih^T + bih for both directions: (320x384, K=256) -> 480 tiles -> 60 blocks
  gemm_bias_kernel<<<(M / 16) * ((3 * GH) / 16) / 8, 256, 0, stream>>>(x, wih_f, bih_f, gi_f, 3 * GH, FDIM);
  gemm_bias_kernel<<<(M / 16) * ((3 * GH) / 16) / 8, 256, 0, stream>>>(x, wih_r, bih_r, gi_r, 3 * GH, FDIM);

  // 4) persistent BiGRU (2 blocks = 2 directions)
  gru_kernel<<<2, 512, 0, stream>>>(gi_f, gi_r, whh_f, whh_r, bhh_f, bhh_r, x2);

  // 5) u = x2 @ Bm^T : (320x256, K=256) -> 320 tiles -> 40 blocks
  gemm_bias_kernel<<<(M / 16) * (MH / 16) / 8, 256, 0, stream>>>(x2, Bm, nullptr, u, MH, 2 * GH);

  // 6) persistent SSM + LN + fc
  mamba_kernel<<<1, 512, 0, stream>>>(Amat, Cm, u, ln_g, ln_b, fc_w, fc_b, out);
}
